// reparameterlizedRNN_22007412425255
// MI455X (gfx1250) — matrix operand order repaired and checked by reading, not compiled, not run on hardware
//
#include <hip/hip_runtime.h>

typedef __attribute__((ext_vector_type(16))) _Float16 v16h;
typedef __attribute__((ext_vector_type(8)))  float    v8f;
typedef __attribute__((ext_vector_type(4)))  float    vf4;

#define N_POP   4
#define RAND_N  512
#define PN      2048      // N_POP * RAND
#define N_F     28
#define N_S     28
#define BATCH   64
#define T_STEPS 256
#define MB      16        // batches per workgroup
#define NWAVE   8
#define KCHUNK  256       // K columns of h owned per wave
#define ALPHA   0.2f

// Branch-free tanh on the native transcendental pipe.
__device__ __forceinline__ float fast_tanh(float x) {
#if __has_builtin(__builtin_amdgcn_tanhf)
    return __builtin_amdgcn_tanhf(x);          // v_tanh_f32 (gfx1250 TRANS op)
#else
    // tanh(x) = 1 - 2/(1 + 2^(2*log2(e)*x));  exact at +/-inf, no branches.
    float e = __builtin_amdgcn_exp2f(x * 2.88539008177792681472f);
    return 1.f - 2.f * __builtin_amdgcn_rcpf(1.f + e);
#endif
}

// ---------------------------------------------------------------------------
// Setup: loading = mu + C @ noise_loading ; split I/U/V/O ; apply softmax(G)/R
// Pack W1[g][0..7]=Vw, [8..11]=Ow, [12..15]=0  (B-matrix of GEMM1, f16)
// Pack B2[g][0..7]=U,  [8..15]=I               (B-matrix of GEMM2, f16)
// ---------------------------------------------------------------------------
__global__ void rnn_setup(const float* __restrict__ G,
                          const float* __restrict__ mu,
                          const float* __restrict__ C,
                          const float* __restrict__ nl,
                          _Float16* __restrict__ w1g,
                          _Float16* __restrict__ b2g)
{
    int g = blockIdx.x * blockDim.x + threadIdx.x;
    if (g >= PN) return;
    int p = g >> 9;

    float g0 = G[0], g1 = G[1], g2 = G[2], g3 = G[3];
    float mx = fmaxf(fmaxf(g0, g1), fmaxf(g2, g3));
    float e0 = __expf(g0 - mx), e1 = __expf(g1 - mx);
    float e2 = __expf(g2 - mx), e3 = __expf(g3 - mx);
    float ep = (p == 0) ? e0 : (p == 1) ? e1 : (p == 2) ? e2 : e3;
    float wp = (ep / (e0 + e1 + e2 + e3)) / (float)RAND_N;

    float acc[N_F];
    const float* nlr = nl + (size_t)g * N_S;
    #pragma unroll 4
    for (int f = 0; f < N_F; ++f) {
        float a = mu[p * N_F + f];
        const float* cr = C + ((size_t)(p * N_F + f)) * N_S;
        #pragma unroll
        for (int s = 0; s < N_S; ++s) a += cr[s] * nlr[s];
        acc[f] = a;
    }
    // I = acc[0..7], U = acc[8..15], V = acc[16..23], O = acc[24..27]
    #pragma unroll
    for (int c = 0; c < 16; ++c) {
        float w1v = (c < 8) ? acc[16 + c] * wp : (c < 12 ? acc[24 + (c - 8)] * wp : 0.f);
        float b2v = (c < 8) ? acc[8 + c] : acc[c - 8];
        w1g[(size_t)g * 16 + c] = (_Float16)w1v;
        b2g[(size_t)g * 16 + c] = (_Float16)b2v;
    }
}

// ---------------------------------------------------------------------------
// Recurrence: one workgroup per 16-batch slice; T-loop resident in LDS.
// ---------------------------------------------------------------------------
__global__ __launch_bounds__(256, 1)
void rnn_recur(const float* __restrict__ u,       // (B, T, 8)
               const float* __restrict__ h0,      // (P*RAND) = 2048
               const float* __restrict__ noise,   // (B, T, 2048)
               const _Float16* __restrict__ w1g,
               const _Float16* __restrict__ b2g,
               float* __restrict__ out)           // (B, T+1, 4)
{
    extern __shared__ char smem[];
    float*    hs   = (float*)smem;                           // 16*2048 f32  (128K)
    _Float16* w1s  = (_Float16*)(smem + 131072);             // 2048*16 f16  (64K)
    _Float16* b2s  = (_Float16*)(smem + 131072 + 65536);     // 2048*16 f16  (64K)
    float*    kred = (float*)(smem + 262144);                // 8*256 f32    (8K)
    float*    kap  = (float*)(smem + 262144 + 8192);         // 16*16 f32    (1K)

    const int tid  = threadIdx.x;
    const int wave = tid >> 5;
    const int lane = tid & 31;
    const int lm   = lane & 15;
    const int hi   = lane >> 4;
    const int b0   = blockIdx.x * MB;

    // Stage packed weights into LDS (uint4 copies)
    {
        const uint4* s1 = (const uint4*)w1g;
        const uint4* s2 = (const uint4*)b2g;
        uint4* d1 = (uint4*)w1s;
        uint4* d2 = (uint4*)b2s;
        for (int i = tid; i < (PN * 16 * 2) / 16; i += 256) { d1[i] = s1[i]; d2[i] = s2[i]; }
    }
    // h_init = broadcast(h0) over batch
    for (int i = tid; i < MB * PN; i += 256) hs[i] = h0[i & (PN - 1)];
    __syncthreads();

    // Persistent B-fragments for GEMM1 (wave's K chunk of [Vw|Ow]) — 64 VGPRs
    v16h b1[8];
    {
        const int kbw = wave * KCHUNK;
        #pragma unroll
        for (int c = 0; c < 8; ++c) {
            int kb = kbw + c * 32 + hi * 8;
            v16h t;
            #pragma unroll
            for (int i = 0; i < 16; ++i) t[i] = w1s[(kb + (i < 8 ? i : i + 8)) * 16 + lm];
            b1[c] = t;
        }
    }

    for (int tt = 0; tt <= T_STEPS; ++tt) {
        // ---- GEMM1: [kappa | y_t] = tanh(h) @ [Vw | Ow], K split over waves
        v8f acc = {0.f, 0.f, 0.f, 0.f, 0.f, 0.f, 0.f, 0.f};
        const int kbw = wave * KCHUNK;
        #pragma unroll
        for (int c = 0; c < 8; ++c) {
            int kb = kbw + c * 32 + hi * 8;
            const vf4* hp0 = (const vf4*)(hs + lm * PN + kb);
            const vf4* hp1 = (const vf4*)(hs + lm * PN + kb + 16);
            vf4 x0 = hp0[0], x1 = hp0[1], y0 = hp1[0], y1 = hp1[1];
            v16h a;
            #pragma unroll
            for (int i = 0; i < 4; ++i) {
                a[i]      = (_Float16)fast_tanh(x0[i]);
                a[i + 4]  = (_Float16)fast_tanh(x1[i]);
                a[i + 8]  = (_Float16)fast_tanh(y0[i]);
                a[i + 12] = (_Float16)fast_tanh(y1[i]);
            }
            acc = __builtin_amdgcn_wmma_f32_16x16x32_f16(false, a, false, b1[c],
                                                         (short)0, acc, false, false);
        }
        // Stage per-wave partial C, then reduce across waves
        #pragma unroll
        for (int vi = 0; vi < 8; ++vi) kred[wave * 256 + vi * 32 + lane] = acc[vi];
        __syncthreads();
        {
            float s = 0.f;
            #pragma unroll
            for (int w = 0; w < NWAVE; ++w) s += kred[w * 256 + tid];
            int vi = tid >> 5, l2 = tid & 31;
            int m = vi + 8 * (l2 >> 4), n = l2 & 15;
            kap[m * 16 + n] = s;                                   // kappa (cols 0..7)
            if (n >= 8 && n < 12)                                  // y_t   (cols 8..11)
                out[((size_t)(b0 + m) * (T_STEPS + 1) + tt) * 4 + (n - 8)] = s;
        }
        __syncthreads();
        if (tt == T_STEPS) break;
        const int t = tt;

        // Prefetch next step's noise slice (global_prefetch_b8), 256B granules
        if (t + 1 < T_STEPS) {
            int pm = tid >> 4, pc = tid & 15;
            const float* pp = noise + ((size_t)(b0 + pm) * T_STEPS + (t + 1)) * PN;
            __builtin_prefetch(pp + pc * 128, 0, 0);
            __builtin_prefetch(pp + pc * 128 + 64, 0, 0);
        }

        // ---- GEMM2: rec+drive = [kappa | u_t] @ [U ; I], K=16 padded to 32
        v16h a2;
        {
            const float* up = u + ((size_t)(b0 + lm) * T_STEPS + t) * 8;
            #pragma unroll
            for (int i = 0; i < 8; ++i) {
                float vv = hi ? up[i] : kap[lm * 16 + i];
                a2[i]     = (_Float16)vv;
                a2[i + 8] = (_Float16)0.f;     // K 16..31 zero pad
            }
        }
        #pragma unroll 4
        for (int j = 0; j < 16; ++j) {
            int jn = (wave * 16 + j) * 16;
            v16h bf;
            #pragma unroll
            for (int i = 0; i < 16; ++i) bf[i] = (_Float16)0.f;
            #pragma unroll
            for (int i = 0; i < 8; ++i) bf[i] = b2s[(size_t)(jn + lm) * 16 + hi * 8 + i];
            v8f c0 = {0.f, 0.f, 0.f, 0.f, 0.f, 0.f, 0.f, 0.f};
            c0 = __builtin_amdgcn_wmma_f32_16x16x32_f16(false, a2, false, bf,
                                                        (short)0, c0, false, false);
            // Fused leaky-integrator update with streamed noise (wave-local cols)
            int n = jn + lm;
            #pragma unroll
            for (int vi = 0; vi < 8; ++vi) {
                int m = vi + 8 * hi;
                float nz = noise[((size_t)(b0 + m) * T_STEPS + t) * PN + n];
                float hv = hs[m * PN + n];
                hs[m * PN + n] = (1.f - ALPHA) * hv + ALPHA * (c0[vi] + nz);
            }
        }
    }
}

// ---------------------------------------------------------------------------
extern "C" void kernel_launch(void* const* d_in, const int* in_sizes, int n_in,
                              void* d_out, int out_size, void* d_ws, size_t ws_size,
                              hipStream_t stream) {
    const float* u  = (const float*)d_in[0];
    const float* G  = (const float*)d_in[1];
    const float* mu = (const float*)d_in[2];
    const float* C  = (const float*)d_in[3];
    const float* h0 = (const float*)d_in[4];
    const float* nl = (const float*)d_in[5];
    const float* nz = (const float*)d_in[6];

    _Float16* w1g = (_Float16*)d_ws;          // 2048*16 f16
    _Float16* b2g = w1g + (size_t)PN * 16;    // 2048*16 f16

    rnn_setup<<<PN / 256, 256, 0, stream>>>(G, mu, C, nl, w1g, b2g);

    size_t lds = 131072 + 65536 + 65536 + 8192 + 1024;   // 271,360 B (<320K/WG)
    rnn_recur<<<BATCH / MB, 256, lds, stream>>>(u, h0, nz, w1g, b2g, (float*)d_out);
}
